// LinearAttentionResidualBlock_76630806495387
// MI455X (gfx1250) — matrix operand order repaired and checked by reading, not compiled, not run on hardware
//
#include <hip/hip_runtime.h>

// ---------------------------------------------------------------------------
// Types for CDNA5 WMMA (gfx1250, wave32)
// ---------------------------------------------------------------------------
typedef __attribute__((ext_vector_type(16))) __bf16 v16bf;
typedef __attribute__((ext_vector_type(8)))  __bf16 v8bf;
typedef __attribute__((ext_vector_type(8)))  float  v8f;
typedef __attribute__((ext_vector_type(4)))  int    v4i;

#if defined(__gfx1250__) && __has_builtin(__builtin_amdgcn_global_load_async_to_lds_b128)
#define USE_ASYNC_LDS 1
#else
#define USE_ASYNC_LDS 0
#endif

#if USE_ASYNC_LDS
__device__ __forceinline__ void async_copy_b128(const void* g, void* l) {
    // toolchain signature: (int4* global_src, int4* lds_dst, imm offset, imm cpol)
    __builtin_amdgcn_global_load_async_to_lds_b128((v4i*)g, (v4i*)l, 0, 0);
}
__device__ __forceinline__ void wait_async0() {
#if __has_builtin(__builtin_amdgcn_s_wait_asynccnt)
    __builtin_amdgcn_s_wait_asynccnt(0);
#else
    asm volatile("s_wait_asynccnt 0x0" ::: "memory");
#endif
}
#endif

__device__ __forceinline__ unsigned short f2bf(float f) {
    unsigned u = __float_as_uint(f);
    u += 0x7FFFu + ((u >> 16) & 1u);          // round-to-nearest-even
    return (unsigned short)(u >> 16);
}
__device__ __forceinline__ float bf2f(unsigned short h) {
    return __uint_as_float(((unsigned)h) << 16);
}
__device__ __forceinline__ float gelu_exact(float x) {
    return 0.5f * x * (1.0f + erff(x * 0.70710678118654752440f));
}

// ---------------------------------------------------------------------------
// fp32 -> bf16 bulk convert (weights)
// ---------------------------------------------------------------------------
__global__ __launch_bounds__(256)
void f32_to_bf16_kernel(const float* __restrict__ in, unsigned short* __restrict__ out, int n) {
    int i = blockIdx.x * 256 + threadIdx.x;
    if (i < n) out[i] = f2bf(in[i]);
}

// ---------------------------------------------------------------------------
// LayerNorm over last dim (C=1024), one block per row, emit bf16
// ---------------------------------------------------------------------------
__global__ __launch_bounds__(256)
void layernorm_bf16_kernel(const float* __restrict__ x, const float* __restrict__ w,
                           const float* __restrict__ b, unsigned short* __restrict__ out, int C) {
    __shared__ float rs[256];
    __shared__ float rs2[256];
    const int row = blockIdx.x;
    const int tid = threadIdx.x;
    const float* xr = x + (size_t)row * C;
    float s = 0.f, s2 = 0.f;
    for (int i = tid; i < C; i += 256) { float v = xr[i]; s += v; s2 += v * v; }
    rs[tid] = s; rs2[tid] = s2;
    __syncthreads();
    for (int off = 128; off > 0; off >>= 1) {
        if (tid < off) { rs[tid] += rs[tid + off]; rs2[tid] += rs2[tid + off]; }
        __syncthreads();
    }
    const float mu   = rs[0] / (float)C;
    const float var  = rs2[0] / (float)C - mu * mu;
    const float rstd = rsqrtf(var + 1e-5f);
    unsigned short* orow = out + (size_t)row * C;
    for (int i = tid; i < C; i += 256)
        orow[i] = f2bf((xr[i] - mu) * rstd * w[i] + b[i]);
}

// ---------------------------------------------------------------------------
// bf16 WMMA GEMM:  Y = X(MxK) @ W(NxK)^T + bias, templated epilogue.
//   Block tile 128x128, K-step 32, 256 threads (8 waves), wave tile 32x64.
//   Double-buffered LDS; tiles staged with GLOBAL_LOAD_ASYNC_TO_LDS_B128
//   (ASYNCcnt-tracked) so the copy for tile t+1 overlaps WMMA on tile t.
// ---------------------------------------------------------------------------
enum { EPI_BF16 = 0, EPI_RESID_F32 = 1, EPI_GELU_BF16 = 2 };

template <int EPI>
__global__ __launch_bounds__(256)
void gemm_bf16_wmma(const unsigned short* __restrict__ X,   // M x K  bf16
                    const unsigned short* __restrict__ W,   // N x K  bf16
                    const float* __restrict__ bias,         // N
                    const float* __restrict__ resid,        // M x N (EPI_RESID_F32)
                    void* __restrict__ Y,
                    int M, int N, int K) {
    constexpr int TK  = 32;
    constexpr int LDK = TK + 8;                 // pad: 80B row stride, kills bank conflicts
    __shared__ alignas(16) unsigned short sX[2][128][LDK];
    __shared__ alignas(16) unsigned short sW[2][128][LDK];

    const int tid   = threadIdx.x;
    const int bm    = blockIdx.x * 128;
    const int bn    = blockIdx.y * 128;
    const int wv    = tid >> 5;
    const int lane  = tid & 31;
    const int half  = lane >> 4;     // 0: lanes 0-15, 1: lanes 16-31
    const int lm    = lane & 15;
    const int waveM = (wv >> 1) * 32;   // 0,32,64,96
    const int waveN = (wv & 1)  * 64;   // 0,64

    // staging: 1024 16B-chunks per (X,W) tile pair; thread owns rows r and r+64
    const int srow = tid >> 2;
    const int scol = (tid & 3) * 8;

    // stage one K-tile (both X and W) into LDS buffer `buf`
    auto stage = [&](int buf, int kt) {
        const size_t gx0 = (size_t)(bm + srow)      * K + kt + scol;
        const size_t gx1 = (size_t)(bm + srow + 64) * K + kt + scol;
        const size_t gw0 = (size_t)(bn + srow)      * K + kt + scol;
        const size_t gw1 = (size_t)(bn + srow + 64) * K + kt + scol;
#if USE_ASYNC_LDS
        async_copy_b128(&X[gx0], &sX[buf][srow][scol]);
        async_copy_b128(&X[gx1], &sX[buf][srow + 64][scol]);
        async_copy_b128(&W[gw0], &sW[buf][srow][scol]);
        async_copy_b128(&W[gw1], &sW[buf][srow + 64][scol]);
#else
        uint4 tx0 = *(const uint4*)&X[gx0];
        uint4 tx1 = *(const uint4*)&X[gx1];
        uint4 tw0 = *(const uint4*)&W[gw0];
        uint4 tw1 = *(const uint4*)&W[gw1];
        *(uint4*)&sX[buf][srow][scol]      = tx0;
        *(uint4*)&sX[buf][srow + 64][scol] = tx1;
        *(uint4*)&sW[buf][srow][scol]      = tw0;
        *(uint4*)&sW[buf][srow + 64][scol] = tw1;
#endif
    };

    v8f acc[2][4];
#pragma unroll
    for (int mi = 0; mi < 2; ++mi)
#pragma unroll
        for (int ni = 0; ni < 4; ++ni)
#pragma unroll
            for (int e = 0; e < 8; ++e) acc[mi][ni][e] = 0.f;

    const int nk = K / TK;

    stage(0, 0);
#if USE_ASYNC_LDS
    wait_async0();
#endif
    __syncthreads();

    for (int t = 0; t < nk; ++t) {
        const int buf = t & 1;
        if (t + 1 < nk) stage(buf ^ 1, (t + 1) * TK);   // overlap with compute below

        // ---- A fragments: 16x32 bf16; lanes 0-15 K[0..7|16..23],
        //      lanes 16-31 K[8..15|24..31] (ISA 7.12.2) ----
        v16bf afrag[2];
#pragma unroll
        for (int mi = 0; mi < 2; ++mi) {
            const int r = waveM + mi * 16 + lm;
            v8bf lo = *(const v8bf*)&sX[buf][r][half * 8];
            v8bf hi = *(const v8bf*)&sX[buf][r][half * 8 + 16];
#pragma unroll
            for (int e = 0; e < 8; ++e) { afrag[mi][e] = lo[e]; afrag[mi][e + 8] = hi[e]; }
        }

        // ---- B fragments: 32x16 bf16; lanes 0-15 K[0..7|16..23], lanes 16-31 K[8..15|24..31] ----
#pragma unroll
        for (int ni = 0; ni < 4; ++ni) {
            const int cc = waveN + ni * 16 + lm;
            v8bf b0 = *(const v8bf*)&sW[buf][cc][half * 8];
            v8bf b1 = *(const v8bf*)&sW[buf][cc][half * 8 + 16];
            v16bf bfrag;
#pragma unroll
            for (int e = 0; e < 8; ++e) { bfrag[e] = b0[e]; bfrag[e + 8] = b1[e]; }
#pragma unroll
            for (int mi = 0; mi < 2; ++mi)
                acc[mi][ni] = __builtin_amdgcn_wmma_f32_16x16x32_bf16(
                    false, afrag[mi], false, bfrag, (short)0, acc[mi][ni], false, false);
        }

#if USE_ASYNC_LDS
        if (t + 1 < nk) wait_async0();   // next tile landed (issued before compute)
#endif
        __syncthreads();                 // everyone done reading buf & next buf ready
    }

    // ---- epilogue: D rows = vgpr_i + 8*half, cols = lane&15 (ISA 7.12.2) ----
#pragma unroll
    for (int mi = 0; mi < 2; ++mi) {
#pragma unroll
        for (int ni = 0; ni < 4; ++ni) {
            const int gr0 = bm + waveM + mi * 16 + half * 8;
            const int gc  = bn + waveN + ni * 16 + lm;
            const float bv = bias[gc];
#pragma unroll
            for (int i = 0; i < 8; ++i) {
                const size_t idx = (size_t)(gr0 + i) * N + gc;
                float y = acc[mi][ni][i] + bv;
                if constexpr (EPI == EPI_RESID_F32) {
                    ((float*)Y)[idx] = y + resid[idx];
                } else if constexpr (EPI == EPI_GELU_BF16) {
                    ((unsigned short*)Y)[idx] = f2bf(gelu_exact(y));
                } else {
                    ((unsigned short*)Y)[idx] = f2bf(y);
                }
            }
        }
    }
}

// ---------------------------------------------------------------------------
// Linear attention core: one workgroup per token.
//   q,k -> elu+1 ; kv[d][e] = sum_h k[h,d] v[h,e] ; qkv = q @ kv ; / (q . ksum)
// ---------------------------------------------------------------------------
__global__ __launch_bounds__(256)
void linear_attention_kernel(const unsigned short* __restrict__ q,
                             const unsigned short* __restrict__ k,
                             const unsigned short* __restrict__ v,
                             unsigned short* __restrict__ attn) {
    __shared__ float sq[1024];
    __shared__ float sk[1024];
    __shared__ float sv[1024];
    __shared__ float skv[64][65];
    __shared__ float sksum[64];
    __shared__ float sqk[16];

    const size_t base = (size_t)blockIdx.x * 1024;
    const int tid = threadIdx.x;

    for (int i = tid; i < 1024; i += 256) {
        float qv = bf2f(q[base + i]);
        float kv_ = bf2f(k[base + i]);
        sq[i] = (qv > 0.f) ? qv + 1.f : __expf(qv);   // elu(x)+1
        sk[i] = (kv_ > 0.f) ? kv_ + 1.f : __expf(kv_);
        sv[i] = bf2f(v[base + i]);
    }
    __syncthreads();

    for (int idx = tid; idx < 4096; idx += 256) {
        const int d = idx >> 6, e = idx & 63;
        float a = 0.f;
#pragma unroll
        for (int h = 0; h < 16; ++h) a += sk[h * 64 + d] * sv[h * 64 + e];
        skv[d][e] = a;
    }
    if (tid < 64) {
        float a = 0.f;
#pragma unroll
        for (int h = 0; h < 16; ++h) a += sk[h * 64 + tid];
        sksum[tid] = a;
    }
    __syncthreads();

    if (tid < 16) {
        float a = 0.f;
#pragma unroll
        for (int d = 0; d < 64; ++d) a += sq[tid * 64 + d] * sksum[d];
        sqk[tid] = a;
    }
    __syncthreads();

    for (int idx = tid; idx < 1024; idx += 256) {
        const int h = idx >> 6, e = idx & 63;
        float a = 0.f;
#pragma unroll
        for (int d = 0; d < 64; ++d) a += sq[h * 64 + d] * skv[d][e];
        attn[base + idx] = f2bf(a / sqk[h]);
    }
}

// ---------------------------------------------------------------------------
// Host-side launch sequence
// ---------------------------------------------------------------------------
extern "C" void kernel_launch(void* const* d_in, const int* in_sizes, int n_in,
                              void* d_out, int out_size, void* d_ws, size_t ws_size,
                              hipStream_t stream) {
    (void)in_sizes; (void)n_in; (void)out_size; (void)ws_size;

    const float* x      = (const float*)d_in[0];
    const float* ln1_w  = (const float*)d_in[1];
    const float* ln1_b  = (const float*)d_in[2];
    const float* q_w    = (const float*)d_in[3];
    const float* q_b    = (const float*)d_in[4];
    const float* k_w    = (const float*)d_in[5];
    const float* k_b    = (const float*)d_in[6];
    const float* v_w    = (const float*)d_in[7];
    const float* v_b    = (const float*)d_in[8];
    const float* o_w    = (const float*)d_in[9];
    const float* o_b    = (const float*)d_in[10];
    const float* ln2_w  = (const float*)d_in[11];
    const float* ln2_b  = (const float*)d_in[12];
    const float* fc_w   = (const float*)d_in[13];
    const float* fc_b   = (const float*)d_in[14];
    const float* proj_w = (const float*)d_in[15];
    const float* proj_b = (const float*)d_in[16];

    const int M = 4 * 4096;   // B*N tokens
    const int C = 1024;
    const int MW = 4096;

    char* ws = (char*)d_ws;
    size_t off = 0;
    auto alloc = [&](size_t bytes) -> char* {
        char* p = ws + off;
        off += (bytes + 255) & ~(size_t)255;
        return p;
    };

    unsigned short* xn_bf    = (unsigned short*)alloc((size_t)M * C * 2);   // reused for xn2
    unsigned short* wq_bf    = (unsigned short*)alloc((size_t)C * C * 2);
    unsigned short* wk_bf    = (unsigned short*)alloc((size_t)C * C * 2);
    unsigned short* wv_bf    = (unsigned short*)alloc((size_t)C * C * 2);
    unsigned short* wo_bf    = (unsigned short*)alloc((size_t)C * C * 2);
    unsigned short* wfc_bf   = (unsigned short*)alloc((size_t)MW * C * 2);
    unsigned short* wproj_bf = (unsigned short*)alloc((size_t)C * MW * 2);
    unsigned short* q_bf     = (unsigned short*)alloc((size_t)M * C * 2);   // 128MB region begins
    unsigned short* k_bf     = (unsigned short*)alloc((size_t)M * C * 2);
    unsigned short* v_bf     = (unsigned short*)alloc((size_t)M * C * 2);
    unsigned short* attn_bf  = (unsigned short*)alloc((size_t)M * C * 2);   // 128MB region ends
    float*          x1       = (float*)alloc((size_t)M * C * 4);
    unsigned short* h_bf     = q_bf;      // alias: q/k/v/attn dead by MLP time
    unsigned short* xn2_bf   = xn_bf;     // alias: LN1 output dead after v GEMM

    // --- weights -> bf16 ---
    f32_to_bf16_kernel<<<(C * C + 255) / 256, 256, 0, stream>>>(q_w,    wq_bf,    C * C);
    f32_to_bf16_kernel<<<(C * C + 255) / 256, 256, 0, stream>>>(k_w,    wk_bf,    C * C);
    f32_to_bf16_kernel<<<(C * C + 255) / 256, 256, 0, stream>>>(v_w,    wv_bf,    C * C);
    f32_to_bf16_kernel<<<(C * C + 255) / 256, 256, 0, stream>>>(o_w,    wo_bf,    C * C);
    f32_to_bf16_kernel<<<(MW * C + 255) / 256, 256, 0, stream>>>(fc_w,   wfc_bf,   MW * C);
    f32_to_bf16_kernel<<<(C * MW + 255) / 256, 256, 0, stream>>>(proj_w, wproj_bf, C * MW);

    // --- LN1 ---
    layernorm_bf16_kernel<<<M, 256, 0, stream>>>(x, ln1_w, ln1_b, xn_bf, C);

    // --- q, k, v projections (WMMA GEMM, bias, bf16 out) ---
    dim3 blk(256);
    dim3 gC(M / 128, C / 128);
    gemm_bf16_wmma<EPI_BF16><<<gC, blk, 0, stream>>>(xn_bf, wq_bf, q_b, nullptr, q_bf, M, C, C);
    gemm_bf16_wmma<EPI_BF16><<<gC, blk, 0, stream>>>(xn_bf, wk_bf, k_b, nullptr, k_bf, M, C, C);
    gemm_bf16_wmma<EPI_BF16><<<gC, blk, 0, stream>>>(xn_bf, wv_bf, v_b, nullptr, v_bf, M, C, C);

    // --- linear attention core (per-token, fp32 VALU: only ~4 GFLOP) ---
    linear_attention_kernel<<<M, 256, 0, stream>>>(q_bf, k_bf, v_bf, attn_bf);

    // --- output projection + residual -> x1 (f32) ---
    gemm_bf16_wmma<EPI_RESID_F32><<<gC, blk, 0, stream>>>(attn_bf, wo_bf, o_b, x, x1, M, C, C);

    // --- LN2 ---
    layernorm_bf16_kernel<<<M, 256, 0, stream>>>(x1, ln2_w, ln2_b, xn2_bf, C);

    // --- MLP: fc + exact GELU -> h (bf16), then proj + residual -> out (f32) ---
    dim3 gFC(M / 128, MW / 128);
    gemm_bf16_wmma<EPI_GELU_BF16><<<gFC, blk, 0, stream>>>(xn2_bf, wfc_bf, fc_b, nullptr, h_bf, M, MW, C);
    gemm_bf16_wmma<EPI_RESID_F32><<<gC, blk, 0, stream>>>(h_bf, wproj_bf, proj_b, x1, (float*)d_out, M, C, MW);
}
